// MultiheadSelfAttention_38225208935024
// MI455X (gfx1250) — matrix operand order repaired and checked by reading, not compiled, not run on hardware
//
#include <hip/hip_runtime.h>

typedef __bf16 bf16;
typedef __attribute__((ext_vector_type(8)))  bf16  v8bf;
typedef __attribute__((ext_vector_type(16))) bf16  v16bf;
typedef __attribute__((ext_vector_type(8)))  float v8f;
typedef __attribute__((ext_vector_type(4)))  int   v4i;

union AFrag { v16bf v; v8bf h[2]; };

#define SCALE 0.125f   // 64^-0.5

// CDNA5 async global->LDS path (ASYNCcnt), with sync fallback.
#if defined(__AMDGCN__) && __has_builtin(__builtin_amdgcn_global_load_async_to_lds_b128) && \
    __has_builtin(__builtin_amdgcn_s_wait_asynccnt)
#define USE_ASYNC_LDS 1
#else
#define USE_ASYNC_LDS 0
#endif

__device__ __forceinline__ void async_copy16(const bf16* g, bf16* l) {
#if USE_ASYNC_LDS
    __builtin_amdgcn_global_load_async_to_lds_b128(
        (__attribute__((address_space(1))) v4i*)(g),
        (__attribute__((address_space(3))) v4i*)(l), 0, 0);
#else
    *(v8bf*)l = *(const v8bf*)g;
#endif
}

__device__ __forceinline__ void wait_async_lds() {
#if USE_ASYNC_LDS
    __builtin_amdgcn_s_wait_asynccnt(0);
#endif
}

// ---------------------------------------------------------------- converts
__global__ void cvt_bf16_kernel(const float* __restrict__ in, bf16* __restrict__ out, int n) {
    int i = blockIdx.x * 256 + threadIdx.x;
    if (i < n) out[i] = (bf16)in[i];
}

__global__ void transpose_bf16_kernel(const float* __restrict__ in, bf16* __restrict__ out,
                                      int rows, int cols) {
    int i = blockIdx.x * 256 + threadIdx.x;
    if (i < rows * cols) {
        int r = i / cols, c = i % cols;
        out[(size_t)c * rows + r] = (bf16)in[i];
    }
}

// ---------------------------------------------------------------- GEMM
// C[M,N] = A[M,K] (row-major bf16) * Bt[N,K]^T (bf16).
// Block: 256 thr / 8 waves; tile 128x128; wave computes 64x32 (4x2 wmma tiles).
template <bool OUT_F32>
__global__ __launch_bounds__(256) void gemm_bf16_kernel(
    const bf16* __restrict__ A, const bf16* __restrict__ Bt,
    bf16* __restrict__ Cb, float* __restrict__ Cf, const float* __restrict__ bias,
    int Mdim, int Ndim, int Kdim) {
    __shared__ __align__(16) bf16 lA[128 * 64];
    __shared__ __align__(16) bf16 lB[128 * 64];

    const int tid  = threadIdx.x;
    const int lane = tid & 31, wid = tid >> 5;
    const int wr = wid >> 2, wc = wid & 3;         // 64-row group, 32-col group
    const int nb = Ndim / 128;
    const int m0 = (blockIdx.x / nb) * 128;
    const int n0 = (blockIdx.x % nb) * 128;
    const int l16 = lane & 15;
    const int kbA = (lane < 16) ? 0 : 8;           // A frag K offset per lane half
    const int kbB = (lane < 16) ? 0 : 8;           // B frag K offset per lane half

    v8f acc[4][2];
    v8f vzero = {};
    #pragma unroll
    for (int i = 0; i < 4; i++)
        #pragma unroll
        for (int j = 0; j < 2; j++) acc[i][j] = vzero;

    const int row  = tid >> 1;                     // 0..127
    const int colh = (tid & 1) * 32;               // 0 or 32

    for (int k0 = 0; k0 < Kdim; k0 += 64) {
        __syncthreads();
        const bf16* gA = A  + (size_t)(m0 + row) * Kdim + k0 + colh;
        const bf16* gB = Bt + (size_t)(n0 + row) * Kdim + k0 + colh;
        #pragma unroll
        for (int i = 0; i < 4; i++) {
            async_copy16(gA + i * 8, &lA[row * 64 + colh + i * 8]);
            async_copy16(gB + i * 8, &lB[row * 64 + colh + i * 8]);
        }
        wait_async_lds();
        __syncthreads();
        if (k0 + 64 < Kdim) {                      // global_prefetch_b8 next K tile
            __builtin_prefetch(gA + 64, 0, 1);
            __builtin_prefetch(gB + 64, 0, 1);
        }
        #pragma unroll
        for (int kk = 0; kk < 64; kk += 32) {
            AFrag af[4], bfr[2];
            #pragma unroll
            for (int mt = 0; mt < 4; mt++) {
                const bf16* p = &lA[(wr * 64 + mt * 16 + l16) * 64 + kk + kbA];
                af[mt].h[0] = *(const v8bf*)p;
                af[mt].h[1] = *(const v8bf*)(p + 16);
            }
            #pragma unroll
            for (int nt = 0; nt < 2; nt++) {
                const bf16* p = &lB[(wc * 32 + nt * 16 + l16) * 64 + kk + kbB];
                bfr[nt].h[0] = *(const v8bf*)p;
                bfr[nt].h[1] = *(const v8bf*)(p + 16);
            }
            #pragma unroll
            for (int mt = 0; mt < 4; mt++)
                #pragma unroll
                for (int nt = 0; nt < 2; nt++)
                    acc[mt][nt] = __builtin_amdgcn_wmma_f32_16x16x32_bf16(
                        false, af[mt].v, false, bfr[nt].v, (short)0, acc[mt][nt],
                        false, false);
        }
    }

    const int halfOff = (lane < 16) ? 0 : 8;
    #pragma unroll
    for (int mt = 0; mt < 4; mt++) {
        #pragma unroll
        for (int nt = 0; nt < 2; nt++) {
            const int n = n0 + wc * 32 + nt * 16 + l16;
            float bv = 0.f;
            if (OUT_F32) bv = bias[n];
            #pragma unroll
            for (int r = 0; r < 8; r++) {
                const int m = m0 + wr * 64 + mt * 16 + r + halfOff;
                const float v = acc[mt][nt][r];
                if (OUT_F32) Cf[(size_t)m * Ndim + n] = v + bv;
                else         Cb[(size_t)m * Ndim + n] = (bf16)v;
            }
        }
    }
}

// ---------------------------------------------------------------- flash attention
// Block: 128 thr / 4 waves, 64 query rows of one (b,h); wave owns a 16-row strip.
__global__ __launch_bounds__(128) void attn_kernel(const bf16* __restrict__ qkv,
                                                   bf16* __restrict__ attno) {
    const int bh = blockIdx.x;                     // 0..63
    const int b  = bh >> 4;
    const int h  = bh & 15;
    const int q0 = blockIdx.y * 64;
    const int tid  = threadIdx.x;
    const int lane = tid & 31, wid = tid >> 5;
    const int l16 = lane & 15;
    const int kbA = (lane < 16) ? 0 : 8;
    const int kbB = (lane < 16) ? 0 : 8;
    const int halfOff = (lane < 16) ? 0 : 8;

    __shared__ __align__(16) bf16 lq[64 * 64];
    __shared__ __align__(16) bf16 lk[64 * 64];
    __shared__ __align__(16) bf16 lvt[64 * 64];    // V transposed: [d][key]
    __shared__ __align__(16) bf16 lp[4][16 * 64];  // per-wave P strip

    const int row  = tid >> 1;                     // 0..63
    const int colh = (tid & 1) * 32;

    { // stage Q (64 x 64) via async global->LDS
        const bf16* g = qkv + (size_t)(b * 2048 + q0 + row) * 3072 + h * 64 + colh;
        #pragma unroll
        for (int i = 0; i < 4; i++)
            async_copy16(g + i * 8, &lq[row * 64 + colh + i * 8]);
        wait_async_lds();
    }
    __syncthreads();

    AFrag qf[2];
    #pragma unroll
    for (int kk = 0; kk < 2; kk++) {
        const bf16* p = &lq[(wid * 16 + l16) * 64 + kk * 32 + kbA];
        qf[kk].h[0] = *(const v8bf*)p;
        qf[kk].h[1] = *(const v8bf*)(p + 16);
    }

    v8f o[4];
    v8f vzero = {};
    float mst[8], lst[8];
    #pragma unroll
    for (int j = 0; j < 4; j++) o[j] = vzero;
    #pragma unroll
    for (int r = 0; r < 8; r++) { mst[r] = -3.0e38f; lst[r] = 0.f; }

    for (int kb0 = 0; kb0 < 2048; kb0 += 64) {
        __syncthreads();                           // lk/lvt reuse guard
        const bf16* gk = qkv + (size_t)(b * 2048 + kb0 + row) * 3072 + 1024 + h * 64 + colh;
        const bf16* gv = qkv + (size_t)(b * 2048 + kb0 + row) * 3072 + 2048 + h * 64 + colh;
        #pragma unroll
        for (int i = 0; i < 4; i++)
            async_copy16(gk + i * 8, &lk[row * 64 + colh + i * 8]);
        v8bf vv[4];
        #pragma unroll
        for (int i = 0; i < 4; i++) vv[i] = *(const v8bf*)(gv + i * 8);
        #pragma unroll
        for (int i = 0; i < 4; i++)
            #pragma unroll
            for (int e = 0; e < 8; e++)
                lvt[(colh + i * 8 + e) * 64 + row] = vv[i][e];
        wait_async_lds();
        __syncthreads();

        // S strip (16 x 64) = Q strip * K^T
        v8f s[4];
        #pragma unroll
        for (int j = 0; j < 4; j++) {
            s[j] = vzero;
            #pragma unroll
            for (int kk = 0; kk < 2; kk++) {
                AFrag kf;
                const bf16* p = &lk[(j * 16 + l16) * 64 + kk * 32 + kbB];
                kf.h[0] = *(const v8bf*)p;
                kf.h[1] = *(const v8bf*)(p + 16);
                s[j] = __builtin_amdgcn_wmma_f32_16x16x32_bf16(
                    false, qf[kk].v, false, kf.v, (short)0, s[j], false, false);
            }
        }

        // online softmax, one row per (VGPR r, lane half)
        #pragma unroll
        for (int r = 0; r < 8; r++) {
            float rm = -3.0e38f;
            #pragma unroll
            for (int j = 0; j < 4; j++) {
                float v = s[j][r] * SCALE;
                s[j][r] = v;
                rm = fmaxf(rm, v);
            }
            #pragma unroll
            for (int off = 1; off < 16; off <<= 1)
                rm = fmaxf(rm, __shfl_xor(rm, off, 32));
            const float nm    = fmaxf(mst[r], rm);
            const float alpha = __expf(mst[r] - nm);
            mst[r] = nm;
            float rs = 0.f;
            #pragma unroll
            for (int j = 0; j < 4; j++) {
                float p = __expf(s[j][r] - nm);
                s[j][r] = p;
                rs += p;
            }
            #pragma unroll
            for (int off = 1; off < 16; off <<= 1)
                rs += __shfl_xor(rs, off, 32);
            lst[r] = lst[r] * alpha + rs;
            #pragma unroll
            for (int j = 0; j < 4; j++) o[j][r] *= alpha;
            const int ml = r + halfOff;
            #pragma unroll
            for (int j = 0; j < 4; j++)
                lp[wid][ml * 64 + j * 16 + l16] = (bf16)s[j][r];
        }

        // O strip += P * V  (P re-read from LDS in A layout; same-wave DS in order)
        #pragma unroll
        for (int kk = 0; kk < 2; kk++) {
            AFrag pf;
            const bf16* pp = &lp[wid][l16 * 64 + kk * 32 + kbA];
            pf.h[0] = *(const v8bf*)pp;
            pf.h[1] = *(const v8bf*)(pp + 16);
            #pragma unroll
            for (int j = 0; j < 4; j++) {
                AFrag vf;
                const bf16* vp = &lvt[(j * 16 + l16) * 64 + kk * 32 + kbB];
                vf.h[0] = *(const v8bf*)vp;
                vf.h[1] = *(const v8bf*)(vp + 16);
                o[j] = __builtin_amdgcn_wmma_f32_16x16x32_bf16(
                    false, pf.v, false, vf.v, (short)0, o[j], false, false);
            }
        }
    }

    #pragma unroll
    for (int r = 0; r < 8; r++) {
        const int ml  = r + halfOff;
        const int mg  = q0 + wid * 16 + ml;
        const float inv = 1.0f / lst[r];
        #pragma unroll
        for (int j = 0; j < 4; j++)
            attno[(size_t)(b * 2048 + mg) * 1024 + h * 64 + j * 16 + l16] =
                (bf16)(o[j][r] * inv);
    }
}

// ---------------------------------------------------------------- launch
extern "C" void kernel_launch(void* const* d_in, const int* in_sizes, int n_in,
                              void* d_out, int out_size, void* d_ws, size_t ws_size,
                              hipStream_t stream) {
    const float* z     = (const float*)d_in[0];   // (4,2048,1024)
    const float* w_qkv = (const float*)d_in[1];   // (1024,3072)
    const float* w_out = (const float*)d_in[2];   // (1024,1024)
    const float* b_out = (const float*)d_in[3];   // (1024,)
    float* out = (float*)d_out;

    char* ws = (char*)d_ws;
    bf16* zb    = (bf16*)(ws + 0);              // 16 MB
    bf16* wqkvT = (bf16*)(ws + 16777216);       //  6 MB  [3072][1024]
    bf16* woutT = (bf16*)(ws + 23068672);       //  2 MB  [1024][1024]
    bf16* qkv   = (bf16*)(ws + 25165824);       // 48 MB  [8192][3072]
    bf16* attno = (bf16*)(ws + 75497472);       // 16 MB  [8192][1024]

    cvt_bf16_kernel<<<(8388608 + 255) / 256, 256, 0, stream>>>(z, zb, 8388608);
    transpose_bf16_kernel<<<(1024 * 3072 + 255) / 256, 256, 0, stream>>>(w_qkv, wqkvT, 1024, 3072);
    transpose_bf16_kernel<<<(1024 * 1024 + 255) / 256, 256, 0, stream>>>(w_out, woutT, 1024, 1024);

    // qkv = z @ w_qkv : M=8192, N=3072, K=1024
    gemm_bf16_kernel<false><<<(8192 / 128) * (3072 / 128), 256, 0, stream>>>(
        zb, wqkvT, qkv, nullptr, nullptr, 8192, 3072, 1024);

    // flash attention: 64 (b,h) x 32 query tiles
    attn_kernel<<<dim3(64, 32), 128, 0, stream>>>(qkv, attno);

    // out = attno @ w_out + b_out : M=8192, N=1024, K=1024
    gemm_bf16_kernel<true><<<(8192 / 128) * (1024 / 128), 256, 0, stream>>>(
        attno, woutT, nullptr, out, b_out, 8192, 1024, 1024);
}
